// RNN_57930518889096
// MI455X (gfx1250) — matrix operand order repaired and checked by reading, not compiled, not run on hardware
//
#include <hip/hip_runtime.h>
#include <math.h>

// ---------------------------------------------------------------------------
// RNN: xproj = X@W1+b1 (big GEMM), then h_t = tanh(xproj_t + h@W2 + b2).
// bf16 WMMA (v_wmma_f32_16x16x32_bf16), f32 accumulate, 2x2 register tiles.
// Recurrence: persistent kernel, W2 slice staged in LDS once (time-invariant),
// agent-scope atomic grid barrier between steps.
// ---------------------------------------------------------------------------

typedef __bf16 bf16_t;
typedef __attribute__((ext_vector_type(16))) __bf16 v16bf;
typedef __attribute__((ext_vector_type(8)))  __bf16 v8bf;
typedef __attribute__((ext_vector_type(4)))  __bf16 v4bf;
typedef __attribute__((ext_vector_type(8)))  float  v8f;

constexpr int Bq = 256;          // batch
constexpr int Tt = 512;          // time
constexpr int In = 512;          // input size
constexpr int Lt = 1024;         // latent size
constexpr int Rr = Bq * Tt;      // 131072 rows of the xproj GEMM
constexpr int NWG = 32;          // persistent grid: one block per nt-pair

__device__ __forceinline__ bf16_t to_bf16(float f) {
    unsigned u = __builtin_bit_cast(unsigned, f);
    unsigned r = u + 0x7FFFu + ((u >> 16) & 1u);   // round-to-nearest-even
    unsigned short h = (unsigned short)(r >> 16);
    return __builtin_bit_cast(bf16_t, h);
}

// --------------------------- prep kernels ----------------------------------

__global__ void __launch_bounds__(256) k_cvt_bf16(const float* __restrict__ s,
                                                  bf16_t* __restrict__ d, int n4) {
    int i = blockIdx.x * blockDim.x + threadIdx.x;
    if (i >= n4) return;
    float4 v = ((const float4*)s)[i];
    v4bf o;
    o[0] = to_bf16(v.x); o[1] = to_bf16(v.y);
    o[2] = to_bf16(v.z); o[3] = to_bf16(v.w);
    ((v4bf*)d)[i] = o;
}

// Repack W [K][1024] f32 into WMMA B-operand lane layout:
// packed[(nt*NKB + kb)*512 + lane*16 + e] = bf16(W[kb*32 + (e>=8)*16 + (lane>=16)*8 + (e&7)][nt*16 + (lane&15)])
__global__ void __launch_bounds__(256) k_pack_B(const float* __restrict__ W,
                                                bf16_t* __restrict__ WP, int K) {
    int tid = blockIdx.x * blockDim.x + threadIdx.x;
    if (tid >= K * Lt) return;
    const int NKB  = K >> 5;
    int e    = tid & 15;
    int lane = (tid >> 4) & 31;
    int rest = tid >> 9;
    int kb   = rest & (NKB - 1);
    int nt   = rest / NKB;
    int n = nt * 16 + (lane & 15);
    int k = kb * 32 + ((e >> 3) << 4) + ((lane >> 4) << 3) + (e & 7);
    WP[tid] = to_bf16(W[(size_t)k * Lt + n]);
}

__global__ void __launch_bounds__(256) k_init(bf16_t* __restrict__ H,
                                              unsigned* __restrict__ bar, int n) {
    int i = blockIdx.x * blockDim.x + threadIdx.x;
    if (i < n) H[i] = __builtin_bit_cast(bf16_t, (unsigned short)0);
    if (i == 0) { bar[0] = 0u; bar[1] = 0u; }
}

// --------------------------- WMMA fragment helpers -------------------------

// A fragment per ISA 16-bit A layout: lane<16 -> K {0..7,16..23}, lane>=16 -> {8..15,24..31}
__device__ __forceinline__ v16bf load_a(const bf16_t* __restrict__ arow, int kb, int ko) {
    v8bf a0 = *(const v8bf*)(arow + kb * 32 + ko);
    v8bf a1 = *(const v8bf*)(arow + kb * 32 + ko + 16);
    return __builtin_shufflevector(a0, a1, 0,1,2,3,4,5,6,7,8,9,10,11,12,13,14,15);
}

__device__ __forceinline__ v8f wmma_bf16(v16bf a, v16bf b, v8f c) {
    return __builtin_amdgcn_wmma_f32_16x16x32_bf16(false, a, false, b, (short)0, c,
                                                   false, false);
}

// --------------------------- xproj GEMM (2x2 tiles/wave) -------------------
// d_out[row, n] = sum_k Xbf[row,k] * W1[k,n] + b1[n]
__global__ void __launch_bounds__(256) k_xproj(const bf16_t* __restrict__ Xbf,
                                               const bf16_t* __restrict__ W1P,
                                               const float* __restrict__ b1,
                                               float* __restrict__ out) {
    constexpr int NKB = In / 32;
    const int lane = threadIdx.x & 31;
    const int wave = (blockIdx.x << 3) + (threadIdx.x >> 5);
    const int mt0 = (wave >> 5) << 1;          // row tiles: pairs of 16
    const int nt0 = (wave & 31) << 1;          // col tiles: pairs of 16
    const int m  = lane & 15;
    const int ko = (lane >> 4) << 3;
    const bf16_t* arow0 = Xbf + ((size_t)mt0 * 16 + m) * In;
    const bf16_t* arow1 = arow0 + (size_t)16 * In;
    const bf16_t* bp0 = W1P + (size_t)nt0 * NKB * 512 + (size_t)lane * 16;
    const bf16_t* bp1 = bp0 + (size_t)NKB * 512;

    v8f acc00 = {0,0,0,0,0,0,0,0}, acc01 = acc00, acc10 = acc00, acc11 = acc00;
#pragma unroll 4
    for (int kb = 0; kb < NKB; ++kb) {
        v16bf a0 = load_a(arow0, kb, ko);
        v16bf a1 = load_a(arow1, kb, ko);
        v16bf b0 = *(const v16bf*)(bp0 + (size_t)kb * 512);
        v16bf b1 = *(const v16bf*)(bp1 + (size_t)kb * 512);
        if (kb + 1 < NKB) {
            __builtin_prefetch((const void*)(bp0 + (size_t)(kb + 1) * 512), 0, 1);
            __builtin_prefetch((const void*)(bp1 + (size_t)(kb + 1) * 512), 0, 1);
        }
        acc00 = wmma_bf16(a0, b0, acc00);
        acc01 = wmma_bf16(a0, b1, acc01);
        acc10 = wmma_bf16(a1, b0, acc10);
        acc11 = wmma_bf16(a1, b1, acc11);
    }

    const int mb = (lane >> 4) << 3;
    v8f accs[4] = {acc00, acc01, acc10, acc11};
#pragma unroll
    for (int q = 0; q < 4; ++q) {
        const int mt = mt0 + (q >> 1);
        const int n  = (nt0 + (q & 1)) * 16 + (lane & 15);
        const float bias = b1[n];
        float* orow = out + (size_t)mt * 16 * Lt + n;
#pragma unroll
        for (int j = 0; j < 8; ++j)
            orow[(size_t)(mb + j) * Lt] = accs[q][j] + bias;
    }
}

// --------------------------- persistent recurrence -------------------------
// 32 WGs x 8 waves: block <-> nt-pair (B slice in LDS, loaded once),
// wave <-> mt-pair. Each wave owns a 32x32 output block (2x2 WMMA tiles).
__global__ void __launch_bounds__(256) k_rnn(bf16_t* __restrict__ Hbuf,
                                             const bf16_t* __restrict__ W2P,
                                             const float* __restrict__ b2,
                                             float* __restrict__ out,
                                             unsigned* __restrict__ bar) {
    constexpr int NKB = Lt / 32;                        // 32 k-blocks
    __shared__ __align__(16) bf16_t ldsB[2 * NKB * 512]; // 64 KB: 2 nt tiles

    const int lane = threadIdx.x & 31;
    const int wv   = threadIdx.x >> 5;                  // 0..7 -> mt pair
    const int nt0  = blockIdx.x << 1;                   // nt pair

    // ---- stage this block's time-invariant W2 slice into LDS (once) ----
    {
        const uint4* src = (const uint4*)(W2P + (size_t)nt0 * NKB * 512);
        uint4* dst = (uint4*)ldsB;
#pragma unroll
        for (int i = 0; i < 16; ++i)                     // 4096 uint4 / 256 thr
            dst[threadIdx.x + 256 * i] = src[threadIdx.x + 256 * i];
    }
    __syncthreads();

    const int m  = lane & 15;
    const int ko = (lane >> 4) << 3;
    const int mb = (lane >> 4) << 3;
    const int mt0 = wv << 1;
    const bf16_t* lb0 = ldsB + (size_t)lane * 16;
    const bf16_t* lb1 = lb0 + (size_t)NKB * 512;
    const size_t HL = (size_t)Bq * Lt;
    unsigned localGen = 0;

    for (int t = 0; t < Tt; ++t) {
        const bf16_t* Hc = Hbuf + (size_t)(t & 1) * HL;
        bf16_t*       Hn = Hbuf + (size_t)((t & 1) ^ 1) * HL;
        const bf16_t* arow0 = Hc + ((size_t)mt0 * 16 + m) * Lt;
        const bf16_t* arow1 = arow0 + (size_t)16 * Lt;

        v8f acc00 = {0,0,0,0,0,0,0,0}, acc01 = acc00, acc10 = acc00, acc11 = acc00;
#pragma unroll 4
        for (int kb = 0; kb < NKB; ++kb) {
            v16bf a0 = load_a(arow0, kb, ko);
            v16bf a1 = load_a(arow1, kb, ko);
            v16bf b0 = *(const v16bf*)(lb0 + (size_t)kb * 512);   // ds_load_b128 x2
            v16bf b1 = *(const v16bf*)(lb1 + (size_t)kb * 512);
            acc00 = wmma_bf16(a0, b0, acc00);
            acc01 = wmma_bf16(a0, b1, acc01);
            acc10 = wmma_bf16(a1, b0, acc10);
            acc11 = wmma_bf16(a1, b1, acc11);
        }

        v8f accs[4] = {acc00, acc01, acc10, acc11};
#pragma unroll
        for (int q = 0; q < 4; ++q) {
            const int mt = mt0 + (q >> 1);
            const int n  = (nt0 + (q & 1)) * 16 + (lane & 15);
            const float bias = b2[n];
#pragma unroll
            for (int j = 0; j < 8; ++j) {
                const int b = mt * 16 + mb + j;
                const size_t oidx = ((size_t)b * Tt + t) * (size_t)Lt + n;
                // xproj value: read once; final h: never re-read from out
                const float xp = __builtin_nontemporal_load(&out[oidx]);
                const float v  = tanhf(accs[q][j] + xp + bias);
                __builtin_nontemporal_store(v, &out[oidx]);
                Hn[(size_t)b * Lt + n] = to_bf16(v);     // stays hot in L2
            }
        }

        // ---- device-wide barrier: all h_t visible before step t+1 ----
        __syncthreads();
        if (threadIdx.x == 0) {
            __builtin_amdgcn_fence(__ATOMIC_RELEASE, "agent");
            unsigned prev = __hip_atomic_fetch_add(&bar[0], 1u, __ATOMIC_RELAXED,
                                                   __HIP_MEMORY_SCOPE_AGENT);
            if (prev == NWG - 1) {
                __hip_atomic_store(&bar[0], 0u, __ATOMIC_RELAXED,
                                   __HIP_MEMORY_SCOPE_AGENT);
                __hip_atomic_fetch_add(&bar[1], 1u, __ATOMIC_RELEASE,
                                       __HIP_MEMORY_SCOPE_AGENT);
            } else {
                while (__hip_atomic_load(&bar[1], __ATOMIC_RELAXED,
                                         __HIP_MEMORY_SCOPE_AGENT) <= localGen)
                    __builtin_amdgcn_s_sleep(1);
            }
            localGen++;
            __builtin_amdgcn_fence(__ATOMIC_ACQUIRE, "agent");
        }
        __syncthreads();
    }
}

// --------------------------- host launcher ---------------------------------
extern "C" void kernel_launch(void* const* d_in, const int* in_sizes, int n_in,
                              void* d_out, int out_size, void* d_ws, size_t ws_size,
                              hipStream_t stream) {
    const float* X  = (const float*)d_in[0];
    // d_in[1] = h0 (zeros; recreated by k_init)
    const float* W1 = (const float*)d_in[2];
    const float* b1 = (const float*)d_in[3];
    const float* W2 = (const float*)d_in[4];
    const float* b2 = (const float*)d_in[5];
    float* out = (float*)d_out;

    // workspace: Xbf 128MB | W1P 1MB | W2P 2MB | H x2 1MB | barrier
    char* ws = (char*)d_ws;
    bf16_t* Xbf  = (bf16_t*)ws;
    bf16_t* W1P  = Xbf  + (size_t)Rr * In;
    bf16_t* W2P  = W1P  + (size_t)In * Lt;
    bf16_t* Hbuf = W2P  + (size_t)Lt * Lt;
    unsigned* bar = (unsigned*)(Hbuf + (size_t)2 * Bq * Lt);

    {   // X -> bf16
        int n4 = (Rr * In) / 4;
        k_cvt_bf16<<<n4 / 256, 256, 0, stream>>>(X, Xbf, n4);
    }
    k_pack_B<<<(In * Lt) / 256, 256, 0, stream>>>(W1, W1P, In);
    k_pack_B<<<(Lt * Lt) / 256, 256, 0, stream>>>(W2, W2P, Lt);
    k_init<<<(2 * Bq * Lt) / 256, 256, 0, stream>>>(Hbuf, bar, 2 * Bq * Lt);
    // xproj: (8192/2 mt-pairs) x (64/2 nt-pairs) waves, 8 waves/block
    k_xproj<<<(Rr / 32) * 32 / 8, 256, 0, stream>>>(Xbf, W1P, b1, out);
    // persistent recurrence
    k_rnn<<<NWG, 256, 0, stream>>>(Hbuf, W2P, b2, out, bar);
}
